// GroupedRouter_87909390614977
// MI455X (gfx1250) — matrix operand order repaired and checked by reading, not compiled, not run on hardware
//
#include <hip/hip_runtime.h>
#include <hip/hip_bf16.h>
#include <stdint.h>

// Problem constants (match reference)
#define B_   4
#define N_   1024
#define D_   2048          // model dim; also H*DH
#define H_   16
#define DH_  128
#define G_   8
#define GSZ_ (N_ / G_)     // 128 tokens per group
#define SROW (N_ + 8)      // padded LDS row stride (floats)

typedef __attribute__((ext_vector_type(16))) __bf16        v16bf;
typedef __attribute__((ext_vector_type(8)))  float         v8f;
typedef __attribute__((ext_vector_type(4)))  unsigned int  v4u;

union Frag { v16bf v; v4u q[2]; };

__device__ __forceinline__ unsigned short f2bf(float f) {
    unsigned u = __float_as_uint(f);
    u += 0x7FFFu + ((u >> 16) & 1u);           // round-to-nearest-even
    return (unsigned short)(u >> 16);
}

// A-fragment (16x32 bf16, M x K): lane<16 -> K[k0..k0+15]
// caller folds the +16 for lanes>=16 into k0. Two 16-byte loads.
__device__ __forceinline__ v16bf load_a_frag(const unsigned short* row, int k0) {
    Frag f;
    f.q[0] = *(const v4u*)(row + k0);
    f.q[1] = *(const v4u*)(row + k0 + 8);
    return f.v;
}

// B-fragment (32x16 bf16, K x N): lane = column, 16 contiguous K values.
// One 32-byte load (two 16B halves).
__device__ __forceinline__ v16bf load_b_frag(const unsigned short* col, int k0) {
    Frag f;
    const v4u* p = (const v4u*)(col + k0);
    f.q[0] = p[0];
    f.q[1] = p[1];
    return f.v;
}

// ---------------------------------------------------------------- converts
__global__ void k_f32_to_bf16(const float* __restrict__ in,
                              unsigned short* __restrict__ out, int n) {
    int i = blockIdx.x * blockDim.x + threadIdx.x;
    if (i < n) out[i] = f2bf(in[i]);
}

// Wq/Wk are [D, H*DH] row-major; store transposed bf16: out[n*D + k] = in[k*D + n]
__global__ void k_transpose_bf16(const float* __restrict__ in,
                                 unsigned short* __restrict__ out) {
    int i = blockIdx.x * blockDim.x + threadIdx.x;
    int nIdx = i / D_;
    int k    = i - nIdx * D_;
    out[(size_t)nIdx * D_ + k] = f2bf(in[(size_t)k * D_ + nIdx]);
}

// ---------------------------------------------- projection GEMM (bf16 WMMA)
// C[M=B*N, 2048] = xb[M, D] @ W   (W given transposed: Wt[n][k])
// Block = 4 waves; each wave computes a 16x64 output tile (4 accumulators).
// Output written as bf16 in [(b*H+h)*N + n]*DH + d layout for the scores GEMM.
__global__ void __launch_bounds__(128)
k_proj_wmma(const unsigned short* __restrict__ xb,
            const unsigned short* __restrict__ Wt,
            unsigned short* __restrict__ outp) {
    const int lane   = threadIdx.x & 31;
    const int wave   = threadIdx.x >> 5;
    const int row0   = blockIdx.y * 16;                 // token tile
    const int col0   = blockIdx.x * 256 + wave * 64;    // feature tile (per wave)
    const int lrow   = lane & 15;
    const int hiHalf = lane >> 4;
    const int ka     = hiHalf ? 16 : 0;                 // A-frag K fold
    const int kb     = hiHalf * 16;                     // B-frag K fold

    v8f zero = {};
    v8f acc[4];
#pragma unroll
    for (int t = 0; t < 4; ++t) acc[t] = zero;

    const unsigned short* arow = xb + (size_t)(row0 + lrow) * D_;
#pragma unroll 4
    for (int kk = 0; kk < D_; kk += 32) {
        v16bf a = load_a_frag(arow, kk + ka);
#pragma unroll
        for (int t = 0; t < 4; ++t) {
            const unsigned short* bcol = Wt + (size_t)(col0 + t * 16 + lrow) * D_;
            v16bf bm = load_b_frag(bcol, kk + kb);
            acc[t] = __builtin_amdgcn_wmma_f32_16x16x32_bf16(
                false, a, false, bm, (short)0, acc[t], false, false);
        }
    }
    // C/D layout: lane = column (lane&15), VGPR j = row j (lanes<16) / j+8
#pragma unroll
    for (int t = 0; t < 4; ++t) {
        int col = col0 + t * 16 + lrow;     // feature index
        int h   = col >> 7;                 // / DH
        int d   = col & (DH_ - 1);
#pragma unroll
        for (int j = 0; j < 8; ++j) {
            int m  = row0 + j + hiHalf * 8;         // global token in [0, B*N)
            int bi = m >> 10;                       // / N
            int n  = m & (N_ - 1);
            outp[((size_t)(bi * H_ + h) * N_ + n) * DH_ + d] = f2bf(acc[t][j]);
        }
    }
}

// ------------------------- fused scores + grouped top-2 mask + softmax
// One block per (b, h, 16-query tile). 4 waves x 16 col-tiles => 16x1024 scores
// kept in LDS, then per-(row,group) threads do max / top-2 / masked softmax.
__global__ void __launch_bounds__(128)
k_scores_softmax(const unsigned short* __restrict__ qb,
                 const unsigned short* __restrict__ kb,
                 float* __restrict__ out) {
    extern __shared__ float smem[];
    float* s_sc   = smem;                       // [16][SROW]
    float* s_gmax = smem + 16 * SROW;           // [16][G]
    float* s_psum = s_gmax + 16 * G_;           // [16][G]

    const int lane   = threadIdx.x & 31;
    const int wave   = threadIdx.x >> 5;
    const int ntile  = blockIdx.x & 63;          // N/16 tiles
    const int bh     = blockIdx.x >> 6;          // b*H + h
    const int bidx   = bh >> 4;
    const int h      = bh & 15;
    const int lrow   = lane & 15;
    const int hiHalf = lane >> 4;
    const int ka     = hiHalf ? 16 : 0;
    const int kfold  = hiHalf * 16;

    const unsigned short* qrow  = qb + ((size_t)bh * N_ + ntile * 16 + lrow) * DH_;
    const unsigned short* kbase = kb + (size_t)bh * N_ * DH_;

    v8f zero = {};
    v8f acc[16];
#pragma unroll
    for (int t = 0; t < 16; ++t) acc[t] = zero;

#pragma unroll
    for (int ks = 0; ks < DH_; ks += 32) {
        v16bf a = load_a_frag(qrow, ks + ka);
#pragma unroll
        for (int t = 0; t < 16; ++t) {
            int m = wave * 256 + t * 16 + lrow;          // key token
            v16bf bm = load_b_frag(kbase + (size_t)m * DH_, ks + kfold);
            acc[t] = __builtin_amdgcn_wmma_f32_16x16x32_bf16(
                false, a, false, bm, (short)0, acc[t], false, false);
        }
    }

    const float scale = 0.0883883476483184f;    // 1/sqrt(128)
#pragma unroll
    for (int t = 0; t < 16; ++t) {
        int col = wave * 256 + t * 16 + lrow;
#pragma unroll
        for (int j = 0; j < 8; ++j) {
            int r = j + hiHalf * 8;
            s_sc[r * SROW + col] = acc[t][j] * scale;
        }
    }
    __syncthreads();

    // thread -> (row, group): consecutive lanes take different rows (bank-friendly)
    const int r = threadIdx.x & 15;
    const int g = threadIdx.x >> 4;
    float* srow = s_sc + r * SROW;

    float gm = -3.402823466e+38f;
#pragma unroll 8
    for (int i = 0; i < GSZ_; ++i) gm = fmaxf(gm, srow[g * GSZ_ + i]);
    s_gmax[r * G_ + g] = gm;
    __syncthreads();

    // top-2 groups (jax top_k tie-break: lowest index wins)
    float gv[G_];
#pragma unroll
    for (int gg = 0; gg < G_; ++gg) gv[gg] = s_gmax[r * G_ + gg];
    float best1 = -3.402823466e+38f; int i1 = 0;
#pragma unroll
    for (int gg = 0; gg < G_; ++gg)
        if (gv[gg] > best1) { best1 = gv[gg]; i1 = gg; }
    float best2 = -3.402823466e+38f; int i2 = -1;
#pragma unroll
    for (int gg = 0; gg < G_; ++gg)
        if (gg != i1 && gv[gg] > best2) { best2 = gv[gg]; i2 = gg; }

    const int   keep   = (g == i1) || (g == i2);
    const float rowmax = best1;                 // max over kept groups == global max

    float ps = 0.0f;
    if (keep) {
        for (int i = 0; i < GSZ_; ++i) {
            float e = __expf(srow[g * GSZ_ + i] - rowmax);
            srow[g * GSZ_ + i] = e;             // cache exp in place (exclusive cells)
            ps += e;
        }
    }
    s_psum[r * G_ + g] = ps;
    __syncthreads();

    float tot = 0.0f;
#pragma unroll
    for (int gg = 0; gg < G_; ++gg) tot += s_psum[r * G_ + gg];
    const float inv = 1.0f / tot;

    float* orow = out + (((size_t)(bidx * N_ + ntile * 16 + r)) * H_ + h) * (size_t)N_;
    for (int i = 0; i < GSZ_; ++i)
        orow[g * GSZ_ + i] = keep ? srow[g * GSZ_ + i] * inv : 0.0f;
}

// ------------------------------------------------------------------ launch
extern "C" void kernel_launch(void* const* d_in, const int* in_sizes, int n_in,
                              void* d_out, int out_size, void* d_ws, size_t ws_size,
                              hipStream_t stream) {
    const float* x  = (const float*)d_in[0];
    const float* Wq = (const float*)d_in[1];
    const float* Wk = (const float*)d_in[2];
    float* out = (float*)d_out;

    char* ws = (char*)d_ws;
    unsigned short* xb  = (unsigned short*)(ws);                        // 16 MiB
    unsigned short* Wqt = (unsigned short*)(ws + ((size_t)16 << 20));   //  8 MiB
    unsigned short* Wkt = (unsigned short*)(ws + ((size_t)24 << 20));   //  8 MiB
    unsigned short* qb  = (unsigned short*)(ws + ((size_t)32 << 20));   // 16 MiB
    unsigned short* kbf = (unsigned short*)(ws + ((size_t)48 << 20));   // 16 MiB

    const int nX = B_ * N_ * D_;            // 8.4M
    k_f32_to_bf16<<<nX / 256, 256, 0, stream>>>(x, xb, nX);
    const int nW = D_ * D_;                 // 4.2M
    k_transpose_bf16<<<nW / 256, 256, 0, stream>>>(Wq, Wqt);
    k_transpose_bf16<<<nW / 256, 256, 0, stream>>>(Wk, Wkt);

    dim3 gProj(D_ / 256, (B_ * N_) / 16);   // 8 x 256 blocks, 128 thr
    k_proj_wmma<<<gProj, 128, 0, stream>>>(xb, Wqt, qb);
    k_proj_wmma<<<gProj, 128, 0, stream>>>(xb, Wkt, kbf);

    const size_t shmem = (size_t)(16 * SROW + 2 * 16 * G_) * sizeof(float); // ~65.5 KB
    k_scores_softmax<<<B_ * H_ * (N_ / 16), 128, shmem, stream>>>(qb, kbf, out);
}
